// Attention_local_46067819217532
// MI455X (gfx1250) — matrix operand order repaired and checked by reading, not compiled, not run on hardware
//
#include <hip/hip_runtime.h>
#include <hip/hip_bf16.h>

// ---------------------------------------------------------------------------
// Problem constants (from reference setup_inputs)
// ---------------------------------------------------------------------------
#define B_    16
#define H_    8
#define T_    196
#define DIM_  512
#define D_    64           // head dim
#define TOPK_ 49
#define MTOK  (B_ * T_)    // 3136 token rows
#define NQKV  (3 * DIM_)   // 1536

typedef __bf16 bf16;
typedef __attribute__((ext_vector_type(8)))  __bf16 v8bf;
typedef __attribute__((ext_vector_type(16))) __bf16 v16bf;
typedef __attribute__((ext_vector_type(8)))  float  v8f;

// Split fp32 into bf16 hi (exact truncation, converts exactly) + bf16 lo (residual).
__device__ __forceinline__ void split_bf16(float x, bf16& hi, bf16& lo) {
  unsigned u  = __builtin_bit_cast(unsigned, x) & 0xFFFF0000u;
  float    fh = __builtin_bit_cast(float, u);
  hi = (bf16)fh;        // exact (low mantissa bits already zero)
  lo = (bf16)(x - fh);  // RNE-rounded residual
}

__device__ __forceinline__ v16bf cat16(v8bf a, v8bf b) {
  return __builtin_shufflevector(a, b, 0,1,2,3,4,5,6,7,8,9,10,11,12,13,14,15);
}

// ---------------------------------------------------------------------------
// Elementwise fp32 -> (bf16 hi, bf16 lo)
// ---------------------------------------------------------------------------
__global__ void k_split(const float* __restrict__ src,
                        bf16* __restrict__ hi, bf16* __restrict__ lo, int n) {
  int i = blockIdx.x * 256 + threadIdx.x;
  if (i < n) split_bf16(src[i], hi[i], lo[i]);
}

// W (K x N, row-major) -> WT hi/lo (N x K, row-major) so that a WMMA B-column
// is a contiguous run of K values. hi plane at [0, N*K), lo plane at +N*K.
__global__ void k_transpose_split(const float* __restrict__ W,
                                  bf16* __restrict__ dstT, int K, int N) {
  int e = blockIdx.x * 256 + threadIdx.x;
  if (e >= K * N) return;
  int n = e / K, k = e - n * K;               // e consecutive -> coalesced writes
  split_bf16(W[(size_t)k * N + n], dstT[e], dstT[(size_t)N * K + e]);
}

// ---------------------------------------------------------------------------
// bf16x3 WMMA GEMM, 16x64 register blocking per wave, compile-time dims:
//   C(MxN,f32) = A(MxK) * B(KxN)
//   A: hi plane row-major at A[0], lo plane at A + M*K      (bf16)
//   B: transposed (NxK); hi at BT[0], lo at BT + N*K        (bf16)
// All plane/column-group strides are constants -> fold into load IOFFSET,
// leaving one VGPR address pair per matrix (no spills).
// QKV==true: scatter the (3136 x 1536) qkv product into Q/K/V [b][h][t][64].
// ---------------------------------------------------------------------------
template <bool QKV, int M, int N, int K>
__global__ __launch_bounds__(256) void
k_gemm(const bf16* __restrict__ A, const bf16* __restrict__ BT,
       float* __restrict__ C0, float* __restrict__ C1, float* __restrict__ C2) {
  const int lane = threadIdx.x & 31;
  const int wave = threadIdx.x >> 5;
  constexpr int stripsN = N >> 6;                  // 64-wide strips
  constexpr int ALO = M * K;                       // element offset of A lo plane
  constexpr int BLO = N * K;                       // element offset of B lo plane
  const int strip = blockIdx.x * 8 + wave;
  if (strip >= (M >> 4) * stripsN) return;
  const int tm  = strip / stripsN;
  const int tn4 = strip - tm * stripsN;

  // ISA 7.12.2 16-bit A 16x32 layout: lane<16 holds K {0..7,16..23} of row lane;
  // lane>=16 holds K {8..15,24..31} of row lane-16.
  const int arow = tm * 16 + (lane & 15);
  const int asub = (lane >> 4) * 8;
  // B 32x16: lanes 0-15 hold K {0..7,16..23} of col lane, lanes 16-31 K {8..15,24..31} of col-16.
  const int bsub = (lane >> 4) * 8;

  const bf16* pA = A  + (size_t)arow * K;                         // hi; lo at +ALO
  const bf16* pB = BT + (size_t)(tn4 * 64 + (lane & 15)) * K;     // g=0 hi; col g at +g*16*K

  v8f acc[4] = {{}, {}, {}, {}};
  for (int k0 = 0; k0 < K; k0 += 32) {
    // prefetch next K tile (global_prefetch_b8)
    __builtin_prefetch(pA + k0 + 32 + asub, 0, 1);
    __builtin_prefetch(pB + k0 + 32 + bsub, 0, 1);

    v16bf ah = cat16(*(const v8bf*)(pA + k0 + asub),
                     *(const v8bf*)(pA + k0 + asub + 16));
    v16bf al = cat16(*(const v8bf*)(pA + ALO + k0 + asub),
                     *(const v8bf*)(pA + ALO + k0 + asub + 16));
#pragma unroll
    for (int g = 0; g < 4; ++g) {
      const bf16* pbg = pB + g * 16 * K + k0 + bsub;   // constant g*16*K folds into IOFFSET
      v16bf bh = cat16(*(const v8bf*)(pbg),       *(const v8bf*)(pbg + 16));
      v16bf bl = cat16(*(const v8bf*)(pbg + BLO), *(const v8bf*)(pbg + BLO + 16));
      // bf16x3: A*B ~= Ahi*Bhi + Ahi*Blo + Alo*Bhi   (f32 accumulate)
      acc[g] = __builtin_amdgcn_wmma_f32_16x16x32_bf16(false, ah, false, bh, (short)0, acc[g], false, false);
      acc[g] = __builtin_amdgcn_wmma_f32_16x16x32_bf16(false, ah, false, bl, (short)0, acc[g], false, false);
      acc[g] = __builtin_amdgcn_wmma_f32_16x16x32_bf16(false, al, false, bh, (short)0, acc[g], false, false);
    }
  }

  // C/D layout: element r of lane l -> M = tm*16 + (l>>4)*8 + r, N = strip col
#pragma unroll
  for (int g = 0; g < 4; ++g) {
    const int nCol = tn4 * 64 + g * 16 + (lane & 15);
    if (QKV) {
      // column of qkv = h*192 + sel*64 + dch, row = b*196 + t
      const int h   = nCol / 192;
      const int j   = nCol - h * 192;
      const int sel = j >> 6;
      const int dch = j & 63;
      float* dst = (sel == 0) ? C0 : (sel == 1) ? C1 : C2;
#pragma unroll
      for (int r = 0; r < 8; ++r) {
        int m = tm * 16 + (lane >> 4) * 8 + r;
        int b = m / T_, t = m - b * T_;
        dst[(((size_t)(b * H_ + h) * T_) + t) * D_ + dch] = acc[g][r];
      }
    } else {
      (void)C1; (void)C2;
#pragma unroll
      for (int r = 0; r < 8; ++r) {
        int m = tm * 16 + (lane >> 4) * 8 + r;
        C0[(size_t)m * N + nCol] = acc[g][r];
      }
    }
  }
}

// ---------------------------------------------------------------------------
// Top-49 of each 196-length adj row. One wave per (b,h,t) row; iterative
// wave-wide argmax (shuffle reduction), clear winner, repeat.
// ---------------------------------------------------------------------------
__global__ void k_topk(const float* __restrict__ adj, int* __restrict__ idx_out) {
  const int lane = threadIdx.x & 31;
  const int wid  = blockIdx.x * 8 + (threadIdx.x >> 5);
  if (wid >= B_ * H_ * T_) return;
  const float* row = adj + (size_t)wid * T_;
  int* out = idx_out + (size_t)wid * TOPK_;

  float vals[7];
#pragma unroll
  for (int s = 0; s < 7; ++s) {
    int j = s * 32 + lane;
    vals[s] = (j < T_) ? row[j] : -INFINITY;
  }

  for (int it = 0; it < TOPK_; ++it) {
    float mv = -INFINITY; int mi = T_;
#pragma unroll
    for (int s = 0; s < 7; ++s) {
      if (vals[s] > mv) { mv = vals[s]; mi = s * 32 + lane; }
    }
#pragma unroll
    for (int off = 16; off; off >>= 1) {
      float ov = __shfl_xor(mv, off, 32);
      int   oi = __shfl_xor(mi, off, 32);
      if (ov > mv || (ov == mv && oi < mi)) { mv = ov; mi = oi; }
    }
    if ((mi & 31) == lane) vals[mi >> 5] = -INFINITY;  // remove winner
    if (lane == 0) out[it] = mi;
  }
}

// ---------------------------------------------------------------------------
// Gathered attention, one wave per (b,h,t) query. Online (flash) softmax over
// the 49 gathered keys; output goes through exact-erf GELU and is written
// directly as bf16 hi/lo in (b*t, 512) layout for the final WMMA GEMM.
// ---------------------------------------------------------------------------
__global__ void k_attn(const float* __restrict__ Q, const float* __restrict__ Km,
                       const float* __restrict__ Vm, const int* __restrict__ IDX,
                       bf16* __restrict__ Ohi, bf16* __restrict__ Olo) {
  const int lane = threadIdx.x & 31;
  const int wid  = blockIdx.x * 8 + (threadIdx.x >> 5);
  if (wid >= B_ * H_ * T_) return;
  const int bh = wid / T_;
  const int t  = wid - bh * T_;

  const float scale = 0.04419417382415922f;  // 512^-0.5 (full dim, per reference)
  const float2 q2 = *(const float2*)(Q + (size_t)wid * D_ + lane * 2);
  const float qx = q2.x * scale, qy = q2.y * scale;
  const int* idxp = IDX + (size_t)wid * TOPK_;

  float m = -INFINITY, s = 0.f, ax = 0.f, ay = 0.f;
  for (int j = 0; j < TOPK_; ++j) {
    const int ki = idxp[j];
    const size_t rbase = ((size_t)bh * T_ + ki) * D_ + lane * 2;
    const float2 kk = *(const float2*)(Km + rbase);
    const float2 vv = *(const float2*)(Vm + rbase);
    float p = qx * kk.x + qy * kk.y;
#pragma unroll
    for (int off = 16; off; off >>= 1) p += __shfl_xor(p, off, 32);
    const float mn = fmaxf(m, p);
    const float cf = expf(m - mn);   // 0 on first iteration (m = -inf)
    const float e  = expf(p - mn);
    s  = s  * cf + e;
    ax = ax * cf + e * vv.x;
    ay = ay * cf + e * vv.y;
    m  = mn;
  }
  const float inv = 1.f / s;
  float ox = ax * inv, oy = ay * inv;
  // exact GELU: 0.5*x*(1+erf(x/sqrt(2)))
  ox = 0.5f * ox * (1.f + erff(ox * 0.70710678118654752f));
  oy = 0.5f * oy * (1.f + erff(oy * 0.70710678118654752f));

  const int b = bh >> 3, h = bh & 7;
  const size_t o = ((size_t)(b * T_ + t)) * DIM_ + h * D_ + lane * 2;
  bf16 h0, l0, h1, l1;
  split_bf16(ox, h0, l0);
  split_bf16(oy, h1, l1);
  Ohi[o] = h0; Ohi[o + 1] = h1;
  Olo[o] = l0; Olo[o + 1] = l1;
}

// ---------------------------------------------------------------------------
// Host-side orchestration
// ---------------------------------------------------------------------------
extern "C" void kernel_launch(void* const* d_in, const int* in_sizes, int n_in,
                              void* d_out, int out_size, void* d_ws, size_t ws_size,
                              hipStream_t stream) {
  (void)in_sizes; (void)n_in; (void)out_size; (void)ws_size;
  const float* x    = (const float*)d_in[0];   // (16,196,512)
  const float* adj  = (const float*)d_in[1];   // (16,8,196,196)
  const float* Wqkv = (const float*)d_in[2];   // (512,1536)
  const float* Wv   = (const float*)d_in[3];   // (512,512)
  float* out = (float*)d_out;                  // (16,196,512)

  // Workspace carve-out (~41 MB total)
  char* w = (char*)d_ws;
  auto carve = [&](size_t bytes) -> void* {
    void* p = (void*)w;
    w += (bytes + 255) & ~(size_t)255;
    return p;
  };
  const size_t nX = (size_t)MTOK * DIM_;               // 1,605,632 elements
  bf16* Xs  = (bf16*)carve(nX * 2 * 2);                // hi plane || lo plane
  bf16* WqT = (bf16*)carve((size_t)NQKV * DIM_ * 2 * 2);
  bf16* WvT = (bf16*)carve((size_t)DIM_ * DIM_ * 2 * 2);
  const size_t nHead = (size_t)B_ * H_ * T_ * D_;      // per Q/K/V
  float* Qb = (float*)carve(nHead * 4);
  float* Kb = (float*)carve(nHead * 4);
  float* Vb = (float*)carve(nHead * 4);
  int* IDX  = (int*)carve((size_t)B_ * H_ * T_ * TOPK_ * 4);
  bf16* Os  = (bf16*)carve(nX * 2 * 2);                // hi plane || lo plane

  // 1) fp32 -> bf16 hi/lo splits and weight transposes
  k_split<<<(int)((nX + 255) / 256), 256, 0, stream>>>(x, Xs, Xs + nX, (int)nX);
  k_transpose_split<<<(DIM_ * NQKV + 255) / 256, 256, 0, stream>>>(Wqkv, WqT, DIM_, NQKV);
  k_transpose_split<<<(DIM_ * DIM_ + 255) / 256, 256, 0, stream>>>(Wv, WvT, DIM_, DIM_);

  // 2) QKV projection: (3136x512)*(512x1536), scatter into Q/K/V [b][h][t][64]
  //    strips = 196 * (1536/64) = 4704 waves -> 588 blocks of 8 waves
  k_gemm<true, MTOK, NQKV, DIM_><<<588, 256, 0, stream>>>(Xs, WqT, Qb, Kb, Vb);

  // 3) Top-49 routing indices: 25088 rows, one wave each
  k_topk<<<3136, 256, 0, stream>>>(adj, IDX);

  // 4) Gathered attention + GELU -> bf16 hi/lo (3136 x 512)
  k_attn<<<3136, 256, 0, stream>>>(Qb, Kb, Vb, IDX, Os, Os + nX);

  // 5) Output projection: (3136x512)*(512x512) -> d_out
  //    strips = 196 * (512/64) = 1568 waves -> 196 blocks
  k_gemm<false, MTOK, DIM_, DIM_><<<196, 256, 0, stream>>>(Os, WvT, out, nullptr, nullptr);
}
